// SingleHead_4724464025789
// MI455X (gfx1250) — matrix operand order repaired and checked by reading, not compiled, not run on hardware
//
#include <hip/hip_runtime.h>
#include <hip/hip_bf16.h>

// ---------------------------------------------------------------------------
// CDNA5 (gfx1250) sparse-causal single-head attention, wave32 + WMMA bf16.
//   B=8, S=2048, D=1024, H=64, k=3 (k read from device scalar).
// Pipeline:
//   k0: Wq/Wk/Wv f32 -> bf16 (workspace)
//   k1: fused QKV projection  (WMMA bf16, f32 accum). Writes Q,K row-major
//       bf16 and V transposed [H][S] bf16 to workspace.
//   k2: flash attention, 1 wave per 16-query tile. S^T = K*Q^T WMMA pairs,
//       analytic sparse-causal mask, online softmax, P*V WMMA.
// ---------------------------------------------------------------------------

typedef __bf16 bf16;
typedef __attribute__((ext_vector_type(16))) __bf16 v16bf;
typedef __attribute__((ext_vector_type(8)))  __bf16 v8bf;
typedef __attribute__((ext_vector_type(8)))  float  v8f;

#define WMMA_BF16(A, B, C) \
  __builtin_amdgcn_wmma_f32_16x16x32_bf16(false, (A), false, (B), (short)0, (C), false, false)

static __device__ __forceinline__ v16bf make_v16(v8bf lo, v8bf hi) {
  v16bf r;
#pragma unroll
  for (int e = 0; e < 8; ++e) { r[e] = lo[e]; r[e + 8] = hi[e]; }
  return r;
}

static __device__ __forceinline__ v8f zero8() {
  v8f z;
#pragma unroll
  for (int e = 0; e < 8; ++e) z[e] = 0.0f;
  return z;
}

// ---------------- kernel 0: weight conversion f32 -> bf16 -------------------
__global__ void wconv_kernel(const float* __restrict__ wq,
                             const float* __restrict__ wk,
                             const float* __restrict__ wv,
                             bf16* __restrict__ out, int n) {
  int t = blockIdx.x * blockDim.x + threadIdx.x;
  if (t < n)            out[t] = (bf16)wq[t];
  else if (t < 2 * n)   out[t] = (bf16)wk[t - n];
  else if (t < 3 * n)   out[t] = (bf16)wv[t - 2 * n];
}

// ---------------- kernel 1: fused QKV projection ----------------------------
// One wave per 16-token tile. M=tokens, N=head dims (64 = 4 N-tiles), K=D.
// A (x tile) converted f32->bf16 inline; B = W row-major bf16 (B[k][n]=W[n][k],
// so lane's column n is a contiguous slice of W's row n).
__global__ __launch_bounds__(32) void proj_kernel(
    const float* __restrict__ x,   // [B*S, D] f32
    const bf16*  __restrict__ Wb,  // [3][H][D] bf16
    bf16* __restrict__ Qb,         // [B*S, H] bf16
    bf16* __restrict__ Kb,         // [B*S, H] bf16
    bf16* __restrict__ Vtb,        // [B][H][S] bf16
    int S, int D, int H) {
  const int lane = threadIdx.x;
  const int hf   = lane >> 4;   // 0 or 1
  const int ln   = lane & 15;
  const int m0   = blockIdx.x * 16;   // global token-row base

  v8f c[3][4];
#pragma unroll
  for (int w = 0; w < 3; ++w)
#pragma unroll
    for (int nt = 0; nt < 4; ++nt) c[w][nt] = zero8();

  const float* xrow = x + (size_t)(m0 + ln) * D;

  for (int d0 = 0; d0 < D; d0 += 32) {
    // A fragment: lane holds row m0+ln; elems 0-7 = K {a..a+7}, 8-15 = K {a+16..a+23}, a=8*hf
    v16bf a;
    const float* p = xrow + d0;
#pragma unroll
    for (int e = 0; e < 8; ++e) a[e]     = (bf16)p[hf * 8 + e];
#pragma unroll
    for (int e = 0; e < 8; ++e) a[e + 8] = (bf16)p[16 + hf * 8 + e];

#pragma unroll
    for (int w = 0; w < 3; ++w) {
#pragma unroll
      for (int nt = 0; nt < 4; ++nt) {
        const bf16* wp = Wb + (size_t)w * H * D + (size_t)(nt * 16 + ln) * D + d0 + hf * 8;
        v16bf bfrag = make_v16(*(const v8bf*)wp, *(const v8bf*)(wp + 16));
        c[w][nt] = WMMA_BF16(a, bfrag, c[w][nt]);
      }
    }
  }

  // C layout: lane holds column n = nt*16+ln, rows M = r + 8*hf.
#pragma unroll
  for (int nt = 0; nt < 4; ++nt) {
#pragma unroll
    for (int r = 0; r < 8; ++r) {
      const int row = m0 + r + 8 * hf;
      const int n   = nt * 16 + ln;
      Qb[(size_t)row * H + n] = (bf16)c[0][nt][r];
      Kb[(size_t)row * H + n] = (bf16)c[1][nt][r];
      const int b = row / S, j = row % S;
      Vtb[((size_t)b * H + n) * S + j] = (bf16)c[2][nt][r];   // transposed V
    }
  }
}

// ---------------- kernel 2: flash attention with sparse causal mask ---------
__global__ __launch_bounds__(32) void attn_kernel(
    const bf16* __restrict__ Qb,   // [B*S, H]
    const bf16* __restrict__ Kb,   // [B*S, H]
    const bf16* __restrict__ Vtb,  // [B][H][S]
    float* __restrict__ out,       // [B*S, H] f32
    const int* __restrict__ kptr,  // sparsity k
    int S, int H, int numQT, float rs) {
  const int lane = threadIdx.x;
  const int hf   = lane >> 4;
  const int ln   = lane & 15;
  const int b    = blockIdx.x / numQT;
  const int qt   = blockIdx.x % numQT;
  const int q0   = qt * 16;
  const int kp1  = kptr[0] + 1;

  // Q as B-fragment of S^T = K*Q^T: lane = query column n=ln, K-dim = head dims.
  v16bf qf[2];
  {
    const bf16* qbase = Qb + (size_t)(b * S + q0 + ln) * H;
#pragma unroll
    for (int f = 0; f < 2; ++f) {
      const bf16* p = qbase + f * 32 + hf * 8;
      qf[f] = make_v16(*(const v8bf*)p, *(const v8bf*)(p + 16));
    }
  }

  v8f acc[4];
#pragma unroll
  for (int c4 = 0; c4 < 4; ++c4) acc[c4] = zero8();
  float m = -__builtin_inff();
  float ssum = 0.0f;

  const int nkt   = qt + 1;          // causal key tiles
  const int iters = (nkt + 1) >> 1;  // 32 keys per iteration
  const int i_q   = q0 + ln;         // this lane's query index (for mask)

  for (int it = 0; it < iters; ++it) {
    const int kbase = it * 32;

    // S^T tiles: M = key (tile of 16), N = query. Accumulate over head dim (2 WMMAs each).
    v8f st[2];
#pragma unroll
    for (int t = 0; t < 2; ++t) {
      const int j = kbase + t * 16 + ln;                 // lane's key row (A layout)
      const bf16* kp = Kb + (size_t)(b * S + j) * H;
      v16bf ka0 = make_v16(*(const v8bf*)(kp + hf * 8),
                           *(const v8bf*)(kp + 16 + hf * 8));
      v16bf ka1 = make_v16(*(const v8bf*)(kp + 32 + hf * 8),
                           *(const v8bf*)(kp + 48 + hf * 8));
      st[t] = WMMA_BF16(ka0, qf[0], zero8());
      st[t] = WMMA_BF16(ka1, qf[1], st[t]);
    }

    // Mask + scale. C layout: lane holds query col i_q, key rows r + 8*hf.
    float lg[2][8];
#pragma unroll
    for (int t = 0; t < 2; ++t)
#pragma unroll
      for (int r = 0; r < 8; ++r) {
        const int j = kbase + t * 16 + r + 8 * hf;
        const int d = i_q - j;
        const bool keep =
            (d >= 0) && ((d <= 2) || (((d % kp1) == 0) && (j > 0 || i_q == 0)));
        lg[t][r] = keep ? st[t][r] * rs : -__builtin_inff();
      }

    // Online softmax: row lives on lane pair (ln, ln+16).
    float lmax = -__builtin_inff();
#pragma unroll
    for (int t = 0; t < 2; ++t)
#pragma unroll
      for (int r = 0; r < 8; ++r) lmax = fmaxf(lmax, lg[t][r]);
    lmax = fmaxf(lmax, __shfl_xor(lmax, 16, 32));
    const float newm  = fmaxf(m, lmax);
    const float alpha = __expf(m - newm);
    m = newm;

    float pr[2][8];
    float lsum = 0.0f;
#pragma unroll
    for (int t = 0; t < 2; ++t)
#pragma unroll
      for (int r = 0; r < 8; ++r) {
        pr[t][r] = __expf(lg[t][r] - newm);
        lsum += pr[t][r];
      }
    lsum += __shfl_xor(lsum, 16, 32);
    ssum = ssum * alpha + lsum;

    // Pack P into bf16 A-fragment: S^T C-layout maps exactly onto A-layout:
    // lanes<16 need keys {0..7,16..23}; lanes>=16 need {8..15,24..31}. Both
    // are precisely st[0]/st[1]'s per-lane rows.
    v16bf pa;
#pragma unroll
    for (int r = 0; r < 8; ++r) { pa[r] = (bf16)pr[0][r]; pa[r + 8] = (bf16)pr[1][r]; }

    // Rescale O accumulators: row q = r + 8*hf lives at lane q.
    float al[8];
#pragma unroll
    for (int r = 0; r < 8; ++r) al[r] = __shfl(alpha, r + 8 * hf, 32);
#pragma unroll
    for (int c4 = 0; c4 < 4; ++c4)
#pragma unroll
      for (int r = 0; r < 8; ++r) acc[c4][r] *= al[r];

    // P * V: B-fragment from transposed V is two 16B loads per lane (keys 8h..8h+7, 16+8h..16+8h+7).
#pragma unroll
    for (int c4 = 0; c4 < 4; ++c4) {
      const bf16* vp = Vtb + ((size_t)b * H + c4 * 16 + ln) * S + kbase + hf * 8;
      v16bf vb = make_v16(*(const v8bf*)vp, *(const v8bf*)(vp + 16));
      acc[c4] = WMMA_BF16(pa, vb, acc[c4]);
    }
  }

  // Epilogue: divide by row sums, store f32.
  float rden[8];
#pragma unroll
  for (int r = 0; r < 8; ++r) rden[r] = 1.0f / __shfl(ssum, r + 8 * hf, 32);
#pragma unroll
  for (int c4 = 0; c4 < 4; ++c4)
#pragma unroll
    for (int r = 0; r < 8; ++r) {
      const int row = b * S + q0 + r + 8 * hf;
      out[(size_t)row * H + c4 * 16 + ln] = acc[c4][r] * rden[r];
    }
}

// ---------------------------------------------------------------------------
extern "C" void kernel_launch(void* const* d_in, const int* in_sizes, int n_in,
                              void* d_out, int out_size, void* d_ws, size_t ws_size,
                              hipStream_t stream) {
  const float* x  = (const float*)d_in[0];
  const float* Wq = (const float*)d_in[1];
  const float* Wk = (const float*)d_in[2];
  const float* Wv = (const float*)d_in[3];
  const int*   kp = (const int*)d_in[4];

  const int B = 8, S = 2048, D = 1024, H = 64;
  const float rs = 0.03125f;  // 1/sqrt(D), D=1024

  char* ws = (char*)d_ws;
  bf16* Wb  = (bf16*)ws;                                   // 3*H*D bf16
  bf16* Qb  = (bf16*)(ws + (size_t)3 * H * D * sizeof(bf16));
  bf16* Kb  = Qb + (size_t)B * S * H;
  bf16* Vtb = Kb + (size_t)B * S * H;

  const int nW = H * D;
  wconv_kernel<<<(3 * nW + 255) / 256, 256, 0, stream>>>(Wq, Wk, Wv, Wb, nW);
  proj_kernel<<<(B * S) / 16, 32, 0, stream>>>(x, Wb, Qb, Kb, Vtb, S, D, H);
  attn_kernel<<<B * (S / 16), 32, 0, stream>>>(Qb, Kb, Vtb, (float*)d_out, kp,
                                               S, H, S / 16, rs);
}
